// MHA__15470472200195
// MI455X (gfx1250) — matrix operand order repaired and checked by reading, not compiled, not run on hardware
//
#include <hip/hip_runtime.h>
#include <hip/hip_bf16.h>

// Problem constants (match reference)
#define NSEQ 4096
#define DMODEL 2048
#define DH 1024

typedef __attribute__((ext_vector_type(16))) _Float16 v16h;
typedef __attribute__((ext_vector_type(8)))  float    v8f;

// ---- async global->LDS support probe (gfx1250 ASYNCcnt path) --------------
#if defined(__has_builtin)
#  if __has_builtin(__builtin_amdgcn_global_load_async_to_lds_b128)
#    define HAVE_ASYNC_LDS 1
#  endif
#endif
#ifndef HAVE_ASYNC_LDS
#  define HAVE_ASYNC_LDS 0
#endif

#if defined(__has_builtin) && __has_builtin(__builtin_amdgcn_s_wait_asynccnt)
#  define WAIT_ASYNC(n) __builtin_amdgcn_s_wait_asynccnt(n)
#else
#  define WAIT_ASYNC(n) asm volatile("s_wait_asynccnt %0" :: "i"(n) : "memory")
#endif

#if HAVE_ASYNC_LDS
// Builtin signature (from hipcc diagnostic): param 0 is
//   'int __attribute__((vector_size(16))) __device__ *'  i.e. AS(1) v4i*,
// so cast both pointers to v4i in their address spaces (const dropped).
typedef int v4i_vs __attribute__((vector_size(16)));
typedef __attribute__((address_space(1))) v4i_vs* gptr_b128;
typedef __attribute__((address_space(3))) v4i_vs* lptr_b128;
#endif

// ---------------------------------------------------------------------------
// f32 -> f16 conversion, 4 elements per thread
// ---------------------------------------------------------------------------
__global__ void cvt_f32_f16_x4(const float* __restrict__ src,
                               _Float16* __restrict__ dst, int n4) {
    int i = blockIdx.x * blockDim.x + threadIdx.x;
    if (i >= n4) return;
    float4 v = ((const float4*)src)[i];
    union { uint2 u; _Float16 h[4]; } t;
    t.h[0] = (_Float16)v.x;
    t.h[1] = (_Float16)v.y;
    t.h[2] = (_Float16)v.z;
    t.h[3] = (_Float16)v.w;
    ((uint2*)dst)[i] = t.u;
}

// ---------------------------------------------------------------------------
// NT GEMM: C[M x Nc] = A[M x K] * Bt[Nc x K]^T, f16 in, f32 accumulate
// (v_wmma_f32_16x16x32_f16).
// Block tile 128x256, K-tile 64. 256 threads = 8 wave32s in a 2x4 grid;
// each wave owns a 64x64 tile = 4x4 WMMA subtiles (32 WMMA per K-tile).
// K-tiles are double-buffered in LDS with the next tile's async
// global_load_async_to_lds_b128 transfers in flight behind the current
// tile's 32 WMMAs (sync: s_wait_asynccnt + workgroup barrier).
// cmode: 0 = f16 row-major C ; 1 = f16 transposed C (b128 stores) ;
//        2 = f32 row-major C
// ---------------------------------------------------------------------------
#define BM 128
#define BN 256
#define BK 64
#define LPAD 72   // LDS row stride in f16 elems (144 B, 16B-aligned rows)

#if HAVE_ASYNC_LDS
__device__ __forceinline__ void issue_tile_async(
    const _Float16* __restrict__ A, int lda,
    const _Float16* __restrict__ Bt, int ldbt,
    _Float16* As_b, _Float16* Bs_b,
    int m0, int n0, int kt, int tid, int arow, int aseg)
{
    // A tile: 128 rows x 64 k -> 4 async b128 per thread
    const _Float16* ga = A + (size_t)(m0 + arow) * lda + kt + aseg;
    _Float16* la = As_b + arow * LPAD + aseg;
#pragma unroll
    for (int c = 0; c < 4; ++c)
        __builtin_amdgcn_global_load_async_to_lds_b128(
            (gptr_b128)(ga + c * 8), (lptr_b128)(la + c * 8), 0, 0);
    // B tile: 256 rows x 64 k -> 8 async b128 per thread (one full row)
    const _Float16* gb = Bt + (size_t)(n0 + tid) * ldbt + kt;
    _Float16* lb = Bs_b + tid * LPAD;
#pragma unroll
    for (int c = 0; c < 8; ++c)
        __builtin_amdgcn_global_load_async_to_lds_b128(
            (gptr_b128)(gb + c * 8), (lptr_b128)(lb + c * 8), 0, 0);
}
#endif

__device__ __forceinline__ void compute_tile(
    const _Float16* As_b, const _Float16* Bs_b,
    v8f acc[4][4], int wm, int wn, int hi, int lm)
{
#pragma unroll
    for (int kh = 0; kh < 2; ++kh) {          // two 32-wide K slices
        const int kb = kh * 32;
        // A fragments: 16x32 f16, ISA layout:
        //   lanes 0-15 : M=lane,    halves 0-7 -> K=0..7,  8-15 -> K=16..23
        //   lanes 16-31: M=lane-16, halves 0-7 -> K=8..15, 8-15 -> K=24..31
        v16h a[4], b[4];
#pragma unroll
        for (int i = 0; i < 4; ++i) {
            const _Float16* p = As_b + (wm * 64 + i * 16 + lm) * LPAD + kb;
            union { uint4 u[2]; v16h v; } t;
            t.u[0] = *(const uint4*)(p + hi * 8);
            t.u[1] = *(const uint4*)(p + 16 + hi * 8);
            a[i] = t.v;
        }
        // B fragments: 32x16 f16: lane = column n (&15),
        //   same K layout as A (lanes 0-15: K=0..7,16..23; 16-31: K=8..15,24..31)
#pragma unroll
        for (int j = 0; j < 4; ++j) {
            const _Float16* p = Bs_b + (wn * 64 + j * 16 + lm) * LPAD + kb;
            union { uint4 u[2]; v16h v; } t;
            t.u[0] = *(const uint4*)(p + hi * 8);
            t.u[1] = *(const uint4*)(p + 16 + hi * 8);
            b[j] = t.v;
        }
#pragma unroll
        for (int i = 0; i < 4; ++i)
#pragma unroll
            for (int j = 0; j < 4; ++j)
                acc[i][j] = __builtin_amdgcn_wmma_f32_16x16x32_f16(
                    false, a[i], false, b[j], (short)0, acc[i][j],
                    false, false);
    }
}

__global__ __launch_bounds__(256) void gemm_nt_f16(
    const _Float16* __restrict__ A, int lda,
    const _Float16* __restrict__ Bt, int ldbt,
    void* __restrict__ C, int ldc,
    int Kdim, int cmode)
{
#if HAVE_ASYNC_LDS
    __shared__ __align__(16) _Float16 As[2][BM * LPAD];   // 2 x 18.4 KB
    __shared__ __align__(16) _Float16 Bs[2][BN * LPAD];   // 2 x 36.9 KB
#else
    __shared__ __align__(16) _Float16 As[1][BM * LPAD];
    __shared__ __align__(16) _Float16 Bs[1][BN * LPAD];
#endif

    const int tid  = threadIdx.x;
    const int lane = tid & 31;
    const int wave = tid >> 5;
    const int wm   = wave >> 2;        // 0..1 : 64-row slab
    const int wn   = wave & 3;         // 0..3 : 64-col slab
    const int m0   = blockIdx.y * BM;
    const int n0   = blockIdx.x * BN;

    const int hi = lane >> 4;          // lane half (K sub-block select)
    const int lm = lane & 15;          // M (A frag) / N (B frag) index

    const int arow = tid >> 1;         // 0..127
    const int aseg = (tid & 1) * 32;   // 0 or 32 (32 f16 = 64B)

    v8f zero = {0.f, 0.f, 0.f, 0.f, 0.f, 0.f, 0.f, 0.f};
    v8f acc[4][4];
#pragma unroll
    for (int i = 0; i < 4; ++i)
#pragma unroll
        for (int j = 0; j < 4; ++j) acc[i][j] = zero;

#if HAVE_ASYNC_LDS
    // -------- async double-buffered pipeline (ASYNCcnt) --------
    issue_tile_async(A, lda, Bt, ldbt, As[0], Bs[0], m0, n0, 0, tid, arow, aseg);
    int buf = 0;
    for (int kt = 0; kt < Kdim; kt += BK, buf ^= 1) {
        if (kt + BK < Kdim) {
            issue_tile_async(A, lda, Bt, ldbt, As[buf ^ 1], Bs[buf ^ 1],
                             m0, n0, kt + BK, tid, arow, aseg);
            WAIT_ASYNC(12);   // 12 newest (next tile) may remain in flight
        } else {
            WAIT_ASYNC(0);
        }
        __syncthreads();      // all waves' current tile visible in LDS
        compute_tile(As[buf], Bs[buf], acc, wm, wn, hi, lm);
        __syncthreads();      // done reading buf before it is refilled
    }
#else
    // -------- synchronous fallback (proven path) --------
    for (int kt = 0; kt < Kdim; kt += BK) {
        __syncthreads();
        {   // A tile: 4x uint4 per thread
            const uint4* g = (const uint4*)(A + (size_t)(m0 + arow) * lda + kt + aseg);
            uint4* s = (uint4*)(As[0] + arow * LPAD + aseg);
            uint4 r0 = g[0], r1 = g[1], r2 = g[2], r3 = g[3];
            s[0] = r0; s[1] = r1; s[2] = r2; s[3] = r3;
        }
        {   // B tile: 8x uint4 per thread (one full row)
            const uint4* g = (const uint4*)(Bt + (size_t)(n0 + tid) * ldbt + kt);
            uint4* s = (uint4*)(Bs[0] + tid * LPAD);
            uint4 r0 = g[0], r1 = g[1], r2 = g[2], r3 = g[3];
            uint4 r4 = g[4], r5 = g[5], r6 = g[6], r7 = g[7];
            s[0] = r0; s[1] = r1; s[2] = r2; s[3] = r3;
            s[4] = r4; s[5] = r5; s[6] = r6; s[7] = r7;
        }
        if (kt + BK < Kdim) {   // hint next K tile (global_prefetch_b8)
            __builtin_prefetch(A  + (size_t)(m0 + arow) * lda  + kt + BK + aseg, 0, 1);
            __builtin_prefetch(Bt + (size_t)(n0 + tid)  * ldbt + kt + BK, 0, 1);
        }
        __syncthreads();
        compute_tile(As[0], Bs[0], acc, wm, wn, hi, lm);
    }
#endif

    // --- epilogue. C/D layout: lane holds column n=lane&15, rows
    //     m = (lane>>4)*8 + r for VGPR r=0..7.
#pragma unroll
    for (int i = 0; i < 4; ++i) {
#pragma unroll
        for (int j = 0; j < 4; ++j) {
            const int m = m0 + wm * 64 + i * 16 + hi * 8;
            const int n = n0 + wn * 64 + j * 16 + lm;
            if (cmode == 2) {
                float* Cf = (float*)C;
#pragma unroll
                for (int r = 0; r < 8; ++r)
                    Cf[(size_t)(m + r) * ldc + n] = acc[i][j][r];
            } else if (cmode == 1) {
                // transposed f16 store: 8 consecutive m -> one b128 store
                _Float16* Ch = (_Float16*)C;
                union { uint4 u; _Float16 h[8]; } t;
#pragma unroll
                for (int r = 0; r < 8; ++r) t.h[r] = (_Float16)acc[i][j][r];
                *(uint4*)&Ch[(size_t)n * ldc + m] = t.u;
            } else {
                _Float16* Ch = (_Float16*)C;
#pragma unroll
                for (int r = 0; r < 8; ++r)
                    Ch[(size_t)(m + r) * ldc + n] = (_Float16)acc[i][j][r];
            }
        }
    }
}

// ---------------------------------------------------------------------------
// out[n, h*DH+e] = sum_d Q[n,d] * T[d,e],  T = K^T V   (no softmax in ref!)
// Pipeline per head h (all NT GEMMs, Bt always transposed-ready):
//   Qh  = xh * Wq^T            (f16, row-major)
//   Kt  = (xh * Wk^T)^T        (f16, transposed store)
//   Vt  = (xh * Wv^T)^T        (f16, transposed store)
//   Tt  = V^T K = Vt * Kt^T    (f16, row-major;  Tt[e][d])
//   O   = Q * Tt^T             (f32 -> d_out columns h*DH..)
// ---------------------------------------------------------------------------
extern "C" void kernel_launch(void* const* d_in, const int* in_sizes, int n_in,
                              void* d_out, int out_size, void* d_ws, size_t ws_size,
                              hipStream_t stream) {
    (void)in_sizes; (void)n_in; (void)out_size; (void)ws_size;

    const float* x = (const float*)d_in[0];
    // d_in order: x, Wq1, Wq2, Wk1, Wk2, Wv1, Wv2
    const float* W[6] = { (const float*)d_in[1], (const float*)d_in[2],
                          (const float*)d_in[3], (const float*)d_in[4],
                          (const float*)d_in[5], (const float*)d_in[6] };
    float* out = (float*)d_out;

    _Float16* ws = (_Float16*)d_ws;
    const size_t ND  = (size_t)NSEQ * DMODEL;   // 8.39M
    const size_t NDH = (size_t)NSEQ * DH;       // 4.19M
    const size_t DD  = (size_t)DH * DH;         // 1.05M
    _Float16* xf = ws;                 // x, f16, (N, D)
    _Float16* wf = xf + ND;            // 6 weights f16 (row-major == Bt!)
    _Float16* q  = wf + 6 * DD;        // per head: (N, DH)
    _Float16* kt = q  + 2 * NDH;       // per head: (DH, N)  transposed K
    _Float16* vt = kt + 2 * NDH;       // per head: (DH, N)  transposed V
    _Float16* tt = vt + 2 * NDH;       // per head: (DH, DH) Tt = V^T K
    // total ws use: ~84 MB of f16

    // ---- converts ----
    {
        int n4 = (int)(ND / 4);
        cvt_f32_f16_x4<<<(n4 + 255) / 256, 256, 0, stream>>>(x, xf, n4);
        int w4 = (int)(DD / 4);
        for (int i = 0; i < 6; ++i)
            cvt_f32_f16_x4<<<(w4 + 255) / 256, 256, 0, stream>>>(
                W[i], wf + (size_t)i * DD, w4);
    }

    dim3 blk(256);
    for (int h = 0; h < 2; ++h) {
        const _Float16* xh  = xf + (size_t)h * DH;        // lda = DMODEL
        const _Float16* wq  = wf + (size_t)(0 + h) * DD;  // Wq1,Wq2
        const _Float16* wk  = wf + (size_t)(2 + h) * DD;  // Wk1,Wk2
        const _Float16* wv  = wf + (size_t)(4 + h) * DD;  // Wv1,Wv2
        _Float16* Qh  = q  + (size_t)h * NDH;
        _Float16* Kth = kt + (size_t)h * NDH;
        _Float16* Vth = vt + (size_t)h * NDH;
        _Float16* Tth = tt + (size_t)h * DD;

        dim3 gQKV(DH / BN, NSEQ / BM);   // (4, 32): Nc=1024, M=4096
        // Q = xh * Wq^T  (row-major f16)
        gemm_nt_f16<<<gQKV, blk, 0, stream>>>(xh, DMODEL, wq, DH,
                                              (void*)Qh, DH, DH, 0);
        // Kt = (xh * Wk^T)^T  (transposed f16 store, ldc = NSEQ)
        gemm_nt_f16<<<gQKV, blk, 0, stream>>>(xh, DMODEL, wk, DH,
                                              (void*)Kth, NSEQ, DH, 1);
        // Vt = (xh * Wv^T)^T
        gemm_nt_f16<<<gQKV, blk, 0, stream>>>(xh, DMODEL, wv, DH,
                                              (void*)Vth, NSEQ, DH, 1);
        // Tt[e][d] = sum_n Vt[e][n]*Kt[d][n]   (M=DH, Nc=DH, K=NSEQ)
        dim3 gT(DH / BN, DH / BM);       // (4, 8)
        gemm_nt_f16<<<gT, blk, 0, stream>>>(Vth, NSEQ, Kth, NSEQ,
                                            (void*)Tth, DH, NSEQ, 0);
        // O[n][e] = sum_d Q[n][d]*Tt[e][d] -> f32 out columns [h*DH, h*DH+DH)
        gemm_nt_f16<<<gQKV, blk, 0, stream>>>(Qh, DH, Tth, DH,
                                              (void*)(out + (size_t)h * DH),
                                              DMODEL, DH, 2);
    }
}
